// MultiHeadedAttention_90520730730612
// MI455X (gfx1250) — matrix operand order repaired and checked by reading, not compiled, not run on hardware
//
#include <hip/hip_runtime.h>

// ---------------------------------------------------------------------------
// Structure-aware multi-head attention for MI455X (gfx1250, wave32, WMMA).
//
// Key algebraic fusions (avoid materializing sk/sv = (B,S,S,64) = 268MB each):
//   q . (struct @ Wsk + bsk)  ==  (q @ Wsk^T) . struct  +  q . bsk
//   sum_k attn*(struct@Wsv+bsv) == (sum_k attn*struct) @ Wsv + bsv
// => structure (126MB) is streamed ONCE, shared by all 12 heads (12 waves in
// one workgroup per 16-row q-tile), flash-style online softmax.
// Structure tiles are double-buffered in LDS via GLOBAL_LOAD_ASYNC_TO_LDS_B128
// (ASYNCcnt), overlapping the roofline-critical HBM stream with WMMA work.
// All large GEMMs use v_wmma_f32_16x16x32_bf16 (f32 accumulate).
// ---------------------------------------------------------------------------

#define B_  4
#define S_  512
#define D_  768
#define H_  12
#define DH_ 64
#define SD_ 30
#define NT_ (S_ / 32)          // 16 k-tiles
#define ALPHA_ 1.0f
#define BETA_  1.0f

typedef __attribute__((ext_vector_type(16))) __bf16 v16bf;
typedef __attribute__((ext_vector_type(8)))  float  v8f;

// A-matrix (16x32 bf16) element e of lane (hi = lane>=16) -> K index
__device__ __forceinline__ int aK(int e, int hi) {
    return 16 * hi + e;
}
// B-matrix (32x16 bf16) element e -> K index
__device__ __forceinline__ int bK(int e, int hi) { return 16 * hi + e; }

// Async global->LDS copy (16B), tracked by ASYNCcnt. Generic LDS pointer's low
// 32 bits are the LDS byte address.
__device__ __forceinline__ void async_copy_b128(const void* g, void* l) {
    unsigned loff = (unsigned)(unsigned long long)l;
    asm volatile("global_load_async_to_lds_b128 %0, %1, off"
                 :: "v"(loff), "v"(g) : "memory");
}

// Issue one structure k-tile (16 q-rows x 32 k x 30 f32 = 61440B) as 3840
// B128 async loads, 10 per thread (384 threads). Per-q-row slabs are
// contiguous in global (k,sd innermost) and 3840B-aligned => 16B alignment ok.
__device__ __forceinline__ void prefetch_struct_tile(
    const float* __restrict__ structure, float* __restrict__ sbuf,
    int b, int q0, int k0, int tid)
{
#pragma unroll
    for (int it = 0; it < 10; ++it) {
        int c = tid + it * 384;            // chunk 0..3839
        int t = c * 16;                    // byte offset in tile
        int qi = t / 3840;                 // q-row (3840B per row)
        int rem = t - qi * 3840;
        const char* g = (const char*)(structure +
                        (((size_t)b * S_ + q0 + qi) * S_ + k0) * SD_) + rem;
        async_copy_b128(g, (char*)sbuf + t);
    }
}

// ---------------------------------------------------------------------------
// QKV projection: out_bf16[b][h][s][d] = ((X @ W)[row][col] + bias[col])*scale
// ---------------------------------------------------------------------------
__global__ void __launch_bounds__(256)
proj_qkv_kernel(const float* __restrict__ X, const float* __restrict__ W,
                const float* __restrict__ bias, __bf16* __restrict__ out,
                float scale)
{
    const int tilesN = D_ / 16;                                   // 48
    int wave = (blockIdx.x * blockDim.x + threadIdx.x) >> 5;
    int lane = threadIdx.x & 31;
    int tm = wave / tilesN, tn = wave % tilesN;
    int m0 = tm * 16, n0 = tn * 16;
    int hi = lane >> 4, ln = lane & 15;
    int mrow = m0 + ln;

    v8f acc = {};
#pragma unroll 1
    for (int k0 = 0; k0 < D_; k0 += 32) {
        v16bf a, bm;
#pragma unroll
        for (int e = 0; e < 16; ++e)
            a[e] = (__bf16)X[(size_t)mrow * D_ + k0 + aK(e, hi)];
#pragma unroll
        for (int e = 0; e < 16; ++e)
            bm[e] = (__bf16)W[(size_t)(k0 + bK(e, hi)) * D_ + n0 + ln];
        acc = __builtin_amdgcn_wmma_f32_16x16x32_bf16(false, a, false, bm,
                                                      (short)0, acc, false, false);
    }
#pragma unroll
    for (int r = 0; r < 8; ++r) {
        int row = m0 + r + 8 * hi;          // token index in [0, B*S)
        int col = n0 + ln;                  // = h*64 + d
        int b = row / S_, s = row % S_;
        int h = col / DH_, d = col % DH_;
        float v = (acc[r] + bias[col]) * scale;
        out[(((size_t)b * H_ + h) * S_ + s) * DH_ + d] = (__bf16)v;
    }
}

// ---------------------------------------------------------------------------
// Output projection: out_f32 = ctx_bf16(B*S, D) @ Wo + bo
// ---------------------------------------------------------------------------
__global__ void __launch_bounds__(256)
proj_out_kernel(const __bf16* __restrict__ X, const float* __restrict__ W,
                const float* __restrict__ bias, float* __restrict__ out)
{
    const int tilesN = D_ / 16;
    int wave = (blockIdx.x * blockDim.x + threadIdx.x) >> 5;
    int lane = threadIdx.x & 31;
    int tm = wave / tilesN, tn = wave % tilesN;
    int m0 = tm * 16, n0 = tn * 16;
    int hi = lane >> 4, ln = lane & 15;
    int mrow = m0 + ln;

    v8f acc = {};
#pragma unroll 1
    for (int k0 = 0; k0 < D_; k0 += 32) {
        v16bf a, bm;
#pragma unroll
        for (int e = 0; e < 16; ++e)
            a[e] = X[(size_t)mrow * D_ + k0 + aK(e, hi)];
#pragma unroll
        for (int e = 0; e < 16; ++e)
            bm[e] = (__bf16)W[(size_t)(k0 + bK(e, hi)) * D_ + n0 + ln];
        acc = __builtin_amdgcn_wmma_f32_16x16x32_bf16(false, a, false, bm,
                                                      (short)0, acc, false, false);
    }
#pragma unroll
    for (int r = 0; r < 8; ++r) {
        int row = m0 + r + 8 * hi;
        int col = n0 + ln;
        out[(size_t)row * D_ + col] = acc[r] + bias[col];
    }
}

// ---------------------------------------------------------------------------
// qs precompute: for each (b,h,q) row of scaled Q:
//   qsw[row][j<30]  = ALPHA * sum_d q[d]*Wsk[j][d]
//   qsw[row][30]    = ALPHA * sum_d q[d]*bsk[d]
// ---------------------------------------------------------------------------
__global__ void __launch_bounds__(256)
qs_kernel(const __bf16* __restrict__ Qb, const float* __restrict__ Wsk,
          const float* __restrict__ bsk, float* __restrict__ qsw)
{
    int row = (blockIdx.x * blockDim.x + threadIdx.x) >> 5; // (b*H+h)*S + s
    int lane = threadIdx.x & 31;
    if (row >= B_ * H_ * S_) return;
    float acc = 0.f;
    if (lane < SD_) {
#pragma unroll
        for (int d = 0; d < DH_; ++d)
            acc += (float)Qb[(size_t)row * DH_ + d] * Wsk[lane * DH_ + d];
        acc *= ALPHA_;
    } else if (lane == SD_) {
#pragma unroll
        for (int d = 0; d < DH_; ++d)
            acc += (float)Qb[(size_t)row * DH_ + d] * bsk[d];
        acc *= ALPHA_;
    }
    qsw[(size_t)row * 32 + lane] = acc;
}

// ---------------------------------------------------------------------------
// Fused attention. Grid: (S/16, B). Block: 384 threads = 12 waves = 12 heads.
// One 16-row q-tile per block; k streamed in 32-wide tiles, online softmax.
// Structure tile double-buffered in LDS via async-to-LDS loads.
// ---------------------------------------------------------------------------
__global__ void __launch_bounds__(384, 1)
attn_kernel(const __bf16* __restrict__ Qb, const __bf16* __restrict__ Kb,
            const __bf16* __restrict__ Vb, const float* __restrict__ qsw,
            const float* __restrict__ structure,
            const unsigned char* __restrict__ mask,
            const float* __restrict__ Wsv, const float* __restrict__ bsv,
            __bf16* __restrict__ ctx_out, float* __restrict__ top_attn)
{
    extern __shared__ char smem[];
    float* s_struct0 = (float*)smem;                 // 16*32*30 f (buf 0)
    float* s_struct1 = s_struct0 + 16 * 32 * SD_;    // 16*32*30 f (buf 1)
    float* s_p    = s_struct1 + 16 * 32 * SD_;       // 12*16*32 f (scores/P/ss)
    float* s_h0   = s_p + H_ * 16 * 32;              // 16*512 f (head-0 raw)
    float* s_wsv  = s_h0 + 16 * S_;                  // 30*64 f
    float* s_bsv  = s_wsv + SD_ * DH_;               // 64 f
    float* s_fac  = s_bsv + DH_;                     // 12*16 f
    float* s_stat = s_fac + H_ * 16;                 // 16*2 f
    unsigned char* s_mask0 = (unsigned char*)(s_stat + 32); // 512 B
    unsigned char* s_mask1 = s_mask0 + 16 * 32;             // 512 B

    const int tid  = threadIdx.x;
    const int h    = tid >> 5;            // wave == head
    const int lane = tid & 31;
    const int hi = lane >> 4, ln = lane & 15;
    const int b  = blockIdx.y;
    const int q0 = blockIdx.x * 16;

    // pre-issue async stage of structure tile 0
    prefetch_struct_tile(structure, s_struct0, b, q0, 0, tid);

    // stage Wsv / bsv once
    for (int i = tid; i < SD_ * DH_; i += 384) s_wsv[i] = Wsv[i];
    if (tid < DH_) s_bsv[tid] = bsv[tid];

    // Q A-fragments (row = q0+ln, K = DH split into 2 chunks of 32)
    const size_t qbase = (((size_t)b * H_ + h) * S_ + q0) * DH_;
    v16bf aq[2];
#pragma unroll
    for (int c = 0; c < 2; ++c)
#pragma unroll
        for (int e = 0; e < 16; ++e)
            aq[c][e] = Qb[qbase + (size_t)ln * DH_ + c * 32 + aK(e, hi)];

    // qs registers for row qi = ln (ALPHA pre-folded)
    const size_t qsbase = (((size_t)b * H_ + h) * S_ + q0 + ln) * 32;
    float qsr[SD_];
#pragma unroll
    for (int j = 0; j < SD_; ++j) qsr[j] = qsw[qsbase + j];
    const float qbdot = qsw[qsbase + SD_];

    float m = -1e30f, lsum = 0.f;
    v8f ctx[4];
#pragma unroll
    for (int dv = 0; dv < 4; ++dv)
#pragma unroll
        for (int r = 0; r < 8; ++r) ctx[dv][r] = 0.f;
    float ssac[15];
#pragma unroll
    for (int j = 0; j < 15; ++j) ssac[j] = 0.f;

    const size_t kvbase = (((size_t)b * H_ + h) * S_) * DH_;

#pragma unroll 1
    for (int kt = 0; kt < NT_; ++kt) {
        const int k0 = kt * 32;
        float*         sst = (kt & 1) ? s_struct1 : s_struct0;
        unsigned char* smk = (kt & 1) ? s_mask1  : s_mask0;

        // prefetch next structure tile into the other buffer (async)
        if (kt + 1 < NT_)
            prefetch_struct_tile(structure, (kt & 1) ? s_struct0 : s_struct1,
                                 b, q0, k0 + 32, tid);
        // mask tile kt (plain loads, tiny)
        for (int i = tid; i < 16 * 32; i += 384) {
            int qi = i >> 5, kk = i & 31;
            smk[i] = mask[((size_t)b * S_ + q0 + qi) * S_ + k0 + kk];
        }
        // wait for tile kt's async loads (next tile's 10/thread may remain)
        if (kt + 1 < NT_) asm volatile("s_wait_asynccnt 0xa" ::: "memory");
        else              asm volatile("s_wait_asynccnt 0x0" ::: "memory");
        __syncthreads();

        // ---- scores: Q(16x64) @ K^T(64x32) via 4 WMMAs, dump raw to LDS ----
#pragma unroll
        for (int kn = 0; kn < 2; ++kn) {
            v8f sacc = {};
            const int key0 = k0 + kn * 16;
#pragma unroll
            for (int c = 0; c < 2; ++c) {
                v16bf kb;
#pragma unroll
                for (int e = 0; e < 16; ++e)
                    kb[e] = Kb[kvbase + (size_t)(key0 + ln) * DH_ + c * 32 + bK(e, hi)];
                sacc = __builtin_amdgcn_wmma_f32_16x16x32_bf16(
                    false, aq[c], false, kb, (short)0, sacc, false, false);
            }
#pragma unroll
            for (int r = 0; r < 8; ++r)
                s_p[(h * 16 + r + 8 * hi) * 32 + kn * 16 + ln] = sacc[r];
        }

        // ---- row scan: lane owns row qi=ln, ki half [16*hi,16*hi+16) ------
        float tmp[16];
        float mloc = -1e30f;
#pragma unroll
        for (int kk = 0; kk < 16; ++kk) {
            int ki = hi * 16 + kk;
            float sv = s_p[(h * 16 + ln) * 32 + ki] + qbdot;
            const float* st = sst + (ln * 32 + ki) * SD_;
#pragma unroll
            for (int j = 0; j < SD_; ++j) sv += qsr[j] * st[j];
            if (smk[ln * 32 + ki]) sv = -1e18f;
            if (h == 0) s_h0[ln * S_ + k0 + ki] = sv;
            tmp[kk] = sv;
            mloc = fmaxf(mloc, sv);
        }
        mloc = fmaxf(mloc, __shfl_xor(mloc, 16, 32));
        float mnew = fmaxf(m, mloc);
        float fac = __expf(m - mnew);
        m = mnew;
        lsum *= fac;
#pragma unroll
        for (int j = 0; j < 15; ++j) ssac[j] *= fac;
#pragma unroll
        for (int kk = 0; kk < 16; ++kk) {
            int ki = hi * 16 + kk;
            float p = __expf(tmp[kk] - mnew);
            lsum += p;
            s_p[(h * 16 + ln) * 32 + ki] = p;
        }
        s_fac[h * 16 + ln] = fac;   // both half-lanes write identical value

        // ---- ss += P . struct (lane: row ln, s' subset of 15) -------------
#pragma unroll
        for (int ki = 0; ki < 32; ++ki) {
            float p = s_p[(h * 16 + ln) * 32 + ki];
            const float* st = sst + (ln * 32 + ki) * SD_;
#pragma unroll
            for (int j = 0; j < 15; ++j) ssac[j] += p * st[hi * 15 + j];
        }

        // ---- rescale ctx accumulators by per-row factor -------------------
        float facr[8];
#pragma unroll
        for (int r = 0; r < 8; ++r) facr[r] = s_fac[h * 16 + r + 8 * hi];
#pragma unroll
        for (int dv = 0; dv < 4; ++dv)
#pragma unroll
            for (int r = 0; r < 8; ++r) ctx[dv][r] *= facr[r];

        // ---- ctx += P(16x32) @ V(32x64) via 4 WMMAs -----------------------
        v16bf pa;
#pragma unroll
        for (int e = 0; e < 16; ++e)
            pa[e] = (__bf16)s_p[(h * 16 + ln) * 32 + aK(e, hi)];
#pragma unroll
        for (int dv = 0; dv < 4; ++dv) {
            v16bf vbf;
#pragma unroll
            for (int e = 0; e < 16; ++e)
                vbf[e] = Vb[kvbase + (size_t)(k0 + bK(e, hi)) * DH_ + dv * 16 + ln];
            ctx[dv] = __builtin_amdgcn_wmma_f32_16x16x32_bf16(
                false, pa, false, vbf, (short)0, ctx[dv], false, false);
        }
        __syncthreads();   // all waves done with sst/smk before re-stage
    }

    // ------------------------------- epilogue ------------------------------
    float ltot = lsum + __shfl_xor(lsum, 16, 32);
    float inv = 1.0f / ltot;
    s_fac[h * 16 + ln] = inv;
    if (h == 0 && hi == 0) { s_stat[ln * 2] = m; s_stat[ln * 2 + 1] = inv; }

    // normalized ss -> LDS slots [qi*32 + s'] (reuse p buffer)
#pragma unroll
    for (int j = 0; j < 15; ++j)
        s_p[(h * 16 + ln) * 32 + hi * 15 + j] = ssac[j] * inv;

    float invr[8];
#pragma unroll
    for (int r = 0; r < 8; ++r) invr[r] = s_fac[h * 16 + r + 8 * hi];

#pragma unroll
    for (int dv = 0; dv < 4; ++dv) {
        int d = dv * 16 + ln;
#pragma unroll
        for (int r = 0; r < 8; ++r) {
            int qi = r + 8 * hi;
            float c2 = s_bsv[d];
            const float* ss = s_p + (h * 16 + qi) * 32;
#pragma unroll
            for (int j = 0; j < SD_; ++j) c2 += ss[j] * s_wsv[j * DH_ + d];
            float val = ctx[dv][r] * invr[r] + BETA_ * c2;
            ctx_out[((size_t)b * S_ + q0 + qi) * D_ + h * DH_ + d] = (__bf16)val;
        }
    }

    // head-0 wave emits the normalized attention map from stored raw scores
    if (h == 0) {
#pragma unroll 1
        for (int qi = 0; qi < 16; ++qi) {
            float mm = s_stat[qi * 2], ii = s_stat[qi * 2 + 1];
            for (int k = lane; k < S_; k += 32)
                top_attn[((size_t)b * S_ + q0 + qi) * S_ + k] =
                    __expf(s_h0[qi * S_ + k] - mm) * ii;
        }
    }
}

// ---------------------------------------------------------------------------
extern "C" void kernel_launch(void* const* d_in, const int* in_sizes, int n_in,
                              void* d_out, int out_size, void* d_ws, size_t ws_size,
                              hipStream_t stream)
{
    (void)in_sizes; (void)n_in; (void)out_size; (void)ws_size;
    const float* key       = (const float*)d_in[0];
    const float* value     = (const float*)d_in[1];
    const float* query     = (const float*)d_in[2];
    const float* structure = (const float*)d_in[3];
    const unsigned char* mask = (const unsigned char*)d_in[4]; // jax bool = 1B
    const float* Wq  = (const float*)d_in[5];
    const float* bq  = (const float*)d_in[6];
    const float* Wk  = (const float*)d_in[7];
    const float* bk  = (const float*)d_in[8];
    const float* Wv  = (const float*)d_in[9];
    const float* bv  = (const float*)d_in[10];
    const float* Wsk = (const float*)d_in[11];
    const float* bsk = (const float*)d_in[12];
    const float* Wsv = (const float*)d_in[13];
    const float* bsv = (const float*)d_in[14];
    const float* Wo  = (const float*)d_in[15];
    const float* bo  = (const float*)d_in[16];

    float* out_f = (float*)d_out;                       // (B,S,D)
    float* top   = out_f + (size_t)B_ * S_ * D_;        // (B,S,S)

    // workspace carve (~15.7 MB total)
    char* ws = (char*)d_ws;
    __bf16* Qb   = (__bf16*)ws;  ws += (size_t)B_ * H_ * S_ * DH_ * 2;
    __bf16* Kb   = (__bf16*)ws;  ws += (size_t)B_ * H_ * S_ * DH_ * 2;
    __bf16* Vb   = (__bf16*)ws;  ws += (size_t)B_ * H_ * S_ * DH_ * 2;
    float*  qsw  = (float*)ws;   ws += (size_t)B_ * H_ * S_ * 32 * 4;
    __bf16* ctxb = (__bf16*)ws;  ws += (size_t)B_ * S_ * D_ * 2;

    const int projBlocks = (B_ * S_ / 16) * (D_ / 16) / 8;   // 768 blocks x 256 thr
    proj_qkv_kernel<<<projBlocks, 256, 0, stream>>>(query, Wq, bq, Qb, 0.125f);
    proj_qkv_kernel<<<projBlocks, 256, 0, stream>>>(key,   Wk, bk, Kb, 1.0f);
    proj_qkv_kernel<<<projBlocks, 256, 0, stream>>>(value, Wv, bv, Vb, 1.0f);

    qs_kernel<<<(B_ * H_ * S_ * 32) / 256, 256, 0, stream>>>(Qb, Wsk, bsk, qsw);

    const size_t smem =
        (size_t)(2 * 16 * 32 * SD_ + H_ * 16 * 32 + 16 * S_ + SD_ * DH_ + DH_ +
                 H_ * 16 + 32) * 4 + 2 * 16 * 32;            // ~186 KB dynamic LDS
    attn_kernel<<<dim3(S_ / 16, B_), 384, smem, stream>>>(
        Qb, Kb, Vb, qsw, structure, mask, Wsv, bsv, ctxb, top);

    proj_out_kernel<<<projBlocks, 256, 0, stream>>>(ctxb, Wo, bo, out_f);
}
